// Decoder_72172630441989
// MI455X (gfx1250) — matrix operand order repaired and checked by reading, not compiled, not run on hardware
//
#include <hip/hip_runtime.h>
#include <math.h>

// Decoder: 63-step LSTM recurrence (sequential k_gates) + batched attention,
// projections, and LM head as WMMA f16 GEMMs (f32 accumulate).
// Weights are pre-converted to f16 once (halves L2 traffic of the 132-GFLOP
// LM head); f16 tiles are staged into LDS by the Tensor Data Mover with
// hardware row padding (LDK=72 halves), synchronized via TENSORcnt.

#define SRC   128
#define BSZ   64
#define HID   512
#define ENCD  512
#define VOC   32000
#define NSTEP 63
#define ROWS  (NSTEP * BSZ)   // 4032

typedef _Float16 h16;
typedef _Float16 v16h __attribute__((ext_vector_type(16)));
typedef _Float16 v8h  __attribute__((ext_vector_type(8)));
typedef float    v8f  __attribute__((ext_vector_type(8)));
typedef unsigned int u32x4 __attribute__((ext_vector_type(4)));
typedef int          i32x8 __attribute__((ext_vector_type(8)));
typedef int          i32x4 __attribute__((ext_vector_type(4)));

static constexpr int KT  = 64;       // K staging chunk (halves) = 128B row
static constexpr int LDK = KT + 8;   // LDS row stride (halves): +16B pad

union AFu { v16h v; v8h h[2]; };

// A fragment, 16x32 (MxK) f16 subtile, row-major in LDS with stride LDK.
// Lanes 0-15 -> M=lane, K {0..7,16..23}; lanes 16-31 -> M=lane-16, K {8..15,24..31}.
__device__ __forceinline__ v16h frag_A(const h16* p0, int lane) {
  const int hl = lane >> 4, l = lane & 15;
  const h16* p = p0 + l * LDK + 8 * hl;
  AFu f;
  f.h[0] = *(const v8h*)(p);
  f.h[1] = *(const v8h*)(p + 16);
  return f.v;
}

// B fragment from B^T stored [N][K] in LDS: lanes 0-15 -> N=lane, K {0..7,16..23};
// lanes 16-31 -> N=lane-16, K {8..15,24..31}.
__device__ __forceinline__ v16h frag_B(const h16* p0, int lane) {
  const int hl = lane >> 4, l = lane & 15;
  const h16* p = p0 + l * LDK + 8 * hl;
  AFu f;
  f.h[0] = *(const v8h*)(p);
  f.h[1] = *(const v8h*)(p + 16);
  return f.v;
}

__device__ __forceinline__ v8f wmma_f16(v16h a, v16h b, v8f c) {
  return __builtin_amdgcn_wmma_f32_16x16x32_f16(false, a, false, b,
                                                (short)0, c, false, false);
}

__device__ __forceinline__ float sigm(float x) { return 1.0f / (1.0f + __expf(-x)); }

// ---------------------------------------------------------------------------
// Tensor Data Mover: DMA a 2-D f16 tile (tile_k halves x tile_rows, global row
// stride row_stride halves) into LDS at lds_byte, adding 16B of padding after
// every 128B row so the LDS layout has stride LDK = 72 halves.
// D# built per CDNA5 ISA sect. 8.
// ---------------------------------------------------------------------------
#if __has_builtin(__builtin_amdgcn_tensor_load_to_lds)
#define HAVE_TDM 1
#else
#define HAVE_TDM 0
#endif

__device__ __forceinline__ unsigned lds_addr_of(const void* p) {
  return (unsigned)(unsigned long long)p;   // low 32 bits = LDS offset
}

#if HAVE_TDM
__device__ __forceinline__ void tdm_load_2d(unsigned lds_byte, const h16* gsrc,
                                            int tile_k, int tile_rows,
                                            int row_stride) {
  unsigned long long ga = (unsigned long long)gsrc;
  u32x4 g0;
  g0[0] = 1u;                                   // count=1 (valid), user mode
  g0[1] = lds_byte;                             // lds_addr
  g0[2] = (unsigned)(ga & 0xffffffffu);         // global_addr[31:0]
  g0[3] = (unsigned)((ga >> 32) & 0x01ffffffu)  // global_addr[56:32]
        | (2u << 30);                           // type = 2 ("image")
  i32x8 g1;
  g1[0] = (1 << 16)      // data_size = 1 -> 2 bytes
        | (1 << 20)      // pad_enable
        | (4 << 22)      // pad_interval code 4 -> every 32 DWORDs (128B)
        | (3 << 25);     // pad_amount  code 3 -> 4 DWORDs (16B)
  g1[1] = (tile_k & 0xffff) << 16;              // tensor_dim0[15:0]
  g1[2] = (tile_rows & 0xffff) << 16;           // tensor_dim1[15:0]
  g1[3] = (tile_k & 0xffff) << 16;              // tile_dim0
  g1[4] = (tile_rows & 0xffff);                 // tile_dim1 (tile_dim2 = 0)
  g1[5] = row_stride;                           // tensor_dim0_stride[31:0]
  g1[6] = 0;
  g1[7] = 0;
  i32x4 z4 = {0, 0, 0, 0};
#if defined(__clang_major__) && (__clang_major__ >= 23)
  i32x8 z8 = {0, 0, 0, 0, 0, 0, 0, 0};
  __builtin_amdgcn_tensor_load_to_lds(g0, g1, z4, z4, z8, 0);
#else
  __builtin_amdgcn_tensor_load_to_lds(g0, g1, z4, z4, 0);
#endif
}
#define TDM_WAIT() __builtin_amdgcn_s_wait_tensorcnt(0)
#else
// Fallback: cooperative f16 copy (256 threads assumed unless nthr given).
#define TDM_WAIT() ((void)0)
#endif

__device__ __forceinline__ void stage_b16_fallback(h16 (*dst)[LDK],
                                                   const h16* gsrc, int rows,
                                                   int row_stride, int tid,
                                                   int nthr) {
  for (int i = tid; i < rows * KT; i += nthr) {
    int r = i >> 6, k = i & 63;
    dst[r][k] = gsrc[(size_t)r * row_stride + k];
  }
}

// ---------------------------------------------------------------------------
// Per-step fused gates GEMM + LSTM pointwise. 8 workgroups; each owns a
// 64-wide hidden slice and all 4 gate slices. gates = [x|h] @ Wg16^T, K=1024.
// Wg16 is the pre-converted f16 interleave [W_ih | W_hh] (2048 x 1024).
// ---------------------------------------------------------------------------
__global__ __launch_bounds__(256) void k_gates(
    const int* __restrict__ tok, const float* __restrict__ emb,
    const h16* __restrict__ Wg16,
    const float* __restrict__ h_in, float* __restrict__ h_out,
    float* __restrict__ c)
{
  __shared__ __align__(16) unsigned char smem[65536];
  h16 (*As)[LDK]      = reinterpret_cast<h16(*)[LDK]>(smem);                 // 64x72
  h16 (*Bs)[64][LDK]  = reinterpret_cast<h16(*)[64][LDK]>(smem + 64*LDK*2);  // 4x64x72
  float (*Gs)[64][64] = reinterpret_cast<float(*)[64][64]>(smem);            // alias, post-K

  const int tid  = threadIdx.x;
  const int lane = tid & 31;
  const int wave = tid >> 5;            // 0..7
  const int j0   = blockIdx.x * 64;     // hidden tile base
  const int g    = wave >> 1;           // gate 0..3 (i,f,g,o)
  const int ns   = (wave & 1) * 32;

  v8f acc[4][2];
  const v8f vz = {0,0,0,0,0,0,0,0};
  for (int m = 0; m < 4; ++m) { acc[m][0] = vz; acc[m][1] = vz; }

  for (int kc = 0; kc < 1024; kc += KT) {
#if HAVE_TDM
    if (wave == 0) {
      #pragma unroll
      for (int gg = 0; gg < 4; ++gg)
        tdm_load_2d(lds_addr_of(&Bs[gg][0][0]),
                    Wg16 + (size_t)(gg * 512 + j0) * 1024 + kc, KT, 64, 1024);
    }
#else
    for (int gg = 0; gg < 4; ++gg)
      stage_b16_fallback(Bs[gg], Wg16 + (size_t)(gg * 512 + j0) * 1024 + kc,
                         64, 1024, tid, 256);
#endif
    // Stage A rows: [x=emb[tok] | h] fp32 -> f16 (gather; not TDM-able)
    for (int i = tid; i < BSZ * KT; i += 256) {
      int b = i >> 6, k = (i & 63) + kc;
      float v = (k < 512) ? emb[(size_t)tok[b] * 512 + k]
                          : h_in[b * 512 + (k - 512)];
      As[b][i & 63] = (h16)v;
    }
    if (wave == 0) TDM_WAIT();
    __syncthreads();
    #pragma unroll
    for (int kk = 0; kk < KT; kk += 32) {
      v16h b0 = frag_B(&Bs[g][ns][kk], lane);
      v16h b1 = frag_B(&Bs[g][ns + 16][kk], lane);
      #pragma unroll
      for (int mt = 0; mt < 4; ++mt) {
        v16h a = frag_A(&As[mt * 16][kk], lane);
        acc[mt][0] = wmma_f16(a, b0, acc[mt][0]);
        acc[mt][1] = wmma_f16(a, b1, acc[mt][1]);
      }
    }
    __syncthreads();
  }

  // Park gate results in LDS (aliases As/Bs, now dead).
  {
    const int hl = lane >> 4, l = lane & 15;
    #pragma unroll
    for (int mt = 0; mt < 4; ++mt)
      #pragma unroll
      for (int r = 0; r < 8; ++r) {
        Gs[g][mt * 16 + r + 8 * hl][ns + l]      = acc[mt][0][r];
        Gs[g][mt * 16 + r + 8 * hl][ns + 16 + l] = acc[mt][1][r];
      }
  }
  __syncthreads();

  // LSTM pointwise (torch gate order i,f,g,o)
  for (int i = tid; i < 64 * 64; i += 256) {
    int b = i >> 6, j = i & 63;
    float ig = sigm(Gs[0][b][j]);
    float fg = sigm(Gs[1][b][j]);
    float gg = tanhf(Gs[2][b][j]);
    float og = sigm(Gs[3][b][j]);
    int idx = b * 512 + j0 + j;
    float cn = fg * c[idx] + ig * gg;
    c[idx] = cn;
    h_out[idx] = og * tanhf(cn);
  }
}

// ---------------------------------------------------------------------------
// Generic WMMA GEMM: Out[m][n] = A[m][:] . Bt16[n][:]   (Bt16 = f16 weight rows)
// M tiled by 64 (blockIdx.y), N tiled by NTILE (blockIdx.x), 8 waves/block.
// B tile (and A tile when AHALF) staged by TDM. EPI: 0=f32, 1=tanh->f16, 2=+bias f32.
// ---------------------------------------------------------------------------
template<int KTOTAL, bool AHALF, int EPI, int NTILE>
__global__ __launch_bounds__(256) void k_gemm(
    const void* __restrict__ Ain, const h16* __restrict__ Bt16,
    const float* __restrict__ bias, void* __restrict__ Out, int ldo)
{
  constexpr int TN = NTILE / 128;   // 16x16 n-tiles per wave (1 or 2)
  __shared__ __align__(16) h16 As[64][LDK];
  __shared__ __align__(16) h16 Bs[NTILE][LDK];

  const int tid  = threadIdx.x;
  const int lane = tid & 31;
  const int wave = tid >> 5;
  const int m0   = blockIdx.y * 64;
  const int n0   = blockIdx.x * NTILE;
  const int ns   = wave * (NTILE / 8);

  v8f acc[4][TN];
  const v8f vz = {0,0,0,0,0,0,0,0};
  for (int m = 0; m < 4; ++m)
    for (int n = 0; n < TN; ++n) acc[m][n] = vz;

  for (int kc = 0; kc < KTOTAL; kc += KT) {
#if HAVE_TDM
    if (wave == 0) {
      tdm_load_2d(lds_addr_of(&Bs[0][0]),
                  Bt16 + (size_t)n0 * KTOTAL + kc, KT, NTILE, KTOTAL);
      if (AHALF)
        tdm_load_2d(lds_addr_of(&As[0][0]),
                    (const h16*)Ain + (size_t)m0 * KTOTAL + kc, KT, 64, KTOTAL);
    }
#else
    stage_b16_fallback(Bs, Bt16 + (size_t)n0 * KTOTAL + kc, NTILE, KTOTAL, tid, 256);
    if (AHALF)
      stage_b16_fallback(As, (const h16*)Ain + (size_t)m0 * KTOTAL + kc,
                         64, KTOTAL, tid, 256);
#endif
    if (!AHALF) {
      for (int i = tid; i < 64 * KT; i += 256) {
        int m = i >> 6, k = i & 63;
        As[m][k] = (h16)((const float*)Ain)[(size_t)(m0 + m) * KTOTAL + kc + k];
      }
    }
    if (wave == 0) TDM_WAIT();
    __syncthreads();
    #pragma unroll
    for (int kk = 0; kk < KT; kk += 32) {
      v16h bf[TN];
      #pragma unroll
      for (int n = 0; n < TN; ++n) bf[n] = frag_B(&Bs[ns + n * 16][kk], lane);
      #pragma unroll
      for (int mt = 0; mt < 4; ++mt) {
        v16h a = frag_A(&As[mt * 16][kk], lane);
        #pragma unroll
        for (int n = 0; n < TN; ++n) acc[mt][n] = wmma_f16(a, bf[n], acc[mt][n]);
      }
    }
    __syncthreads();
  }

  const int hl = lane >> 4, l = lane & 15;
  #pragma unroll
  for (int mt = 0; mt < 4; ++mt)
    #pragma unroll
    for (int n = 0; n < TN; ++n) {
      int col = n0 + ns + n * 16 + l;
      float bv = (EPI == 2) ? bias[col] : 0.0f;
      #pragma unroll
      for (int r = 0; r < 8; ++r) {
        int row = m0 + mt * 16 + r + 8 * hl;
        float v = acc[mt][n][r];
        if (EPI == 1)
          ((h16*)Out)[(size_t)row * ldo + col] = (h16)tanhf(v);
        else
          ((float*)Out)[(size_t)row * ldo + col] = v + bv;
      }
    }
}

// ---------------------------------------------------------------------------
// Batched attention: one workgroup (128 threads, thread == source position s)
// per (t, b). scores -> masked softmax over s -> ctx; writes [ctx|h] as f16.
// ---------------------------------------------------------------------------
__global__ __launch_bounds__(128) void k_attn(
    const float* __restrict__ Q, const float* __restrict__ enc,
    const unsigned char* __restrict__ mask, const float* __restrict__ H,
    h16* __restrict__ czh)
{
  const int b = blockIdx.x;
  const int r = blockIdx.y * BSZ + b;   // global row t*64+b
  const int s = threadIdx.x;            // 0..127

  __shared__ float qs[ENCD];
  __shared__ float red[SRC];
  __shared__ float attn[SRC];

  for (int e = s; e < ENCD; e += 128) qs[e] = Q[(size_t)r * ENCD + e];
  __syncthreads();

  const float4* er4 = (const float4*)(enc + ((size_t)s * BSZ + b) * ENCD);
  const float4* qs4 = (const float4*)qs;
  float sc = 0.f;
  #pragma unroll 4
  for (int e = 0; e < ENCD / 4; ++e) {
    float4 v = er4[e], w = qs4[e];
    sc += v.x * w.x + v.y * w.y + v.z * w.z + v.w * w.w;
  }
  const bool msk = mask[s * BSZ + b] != 0;
  if (msk) sc = -3.0e38f;

  red[s] = sc; __syncthreads();
  for (int off = 64; off > 0; off >>= 1) {
    if (s < off) red[s] = fmaxf(red[s], red[s + off]);
    __syncthreads();
  }
  const float mx = red[0]; __syncthreads();
  float ex = msk ? 0.f : __expf(sc - mx);
  red[s] = ex; __syncthreads();
  for (int off = 64; off > 0; off >>= 1) {
    if (s < off) red[s] += red[s + off];
    __syncthreads();
  }
  const float inv = 1.0f / red[0];
  attn[s] = ex * inv; __syncthreads();

  float cacc[4] = {0.f, 0.f, 0.f, 0.f};
  for (int ss = 0; ss < SRC; ++ss) {
    float a = attn[ss];
    const float* erow = enc + ((size_t)ss * BSZ + b) * ENCD;
    #pragma unroll
    for (int u = 0; u < 4; ++u) cacc[u] += a * erow[s + u * 128];
  }
  h16* dst = czh + (size_t)r * (ENCD + HID);
  #pragma unroll
  for (int u = 0; u < 4; ++u) dst[s + u * 128] = (h16)cacc[u];
  for (int e = s; e < HID; e += 128)
    dst[ENCD + e] = (h16)H[(size_t)r * HID + e];
}

// ---------------------------------------------------------------------------
// One-time weight conversions fp32 -> f16 (deterministic, run every call).
// ---------------------------------------------------------------------------
__global__ __launch_bounds__(256) void k_cvt(const float* __restrict__ src,
                                             h16* __restrict__ dst, int n) {
  for (int i = blockIdx.x * 256 + threadIdx.x; i < n; i += gridDim.x * 256)
    dst[i] = (h16)src[i];
}

__global__ __launch_bounds__(256) void k_cvt_gates(
    const float* __restrict__ wih, const float* __restrict__ whh,
    h16* __restrict__ dst) {           // dst [2048][1024] = [W_ih | W_hh]
  for (int i = blockIdx.x * 256 + threadIdx.x; i < 2048 * 1024;
       i += gridDim.x * 256) {
    int n = i >> 10, k = i & 1023;
    dst[i] = (h16)((k < 512) ? wih[(size_t)n * 512 + k]
                             : whh[(size_t)n * 512 + (k - 512)]);
  }
}

// ---------------------------------------------------------------------------
extern "C" void kernel_launch(void* const* d_in, const int* in_sizes, int n_in,
                              void* d_out, int out_size, void* d_ws, size_t ws_size,
                              hipStream_t stream)
{
  (void)in_sizes; (void)n_in; (void)out_size; (void)ws_size;

  const int*           targets = (const int*)d_in[0];
  const unsigned char* mask    = (const unsigned char*)d_in[1];  // numpy bool = 1B
  const float*         enc     = (const float*)d_in[2];
  const float*         h0      = (const float*)d_in[3];
  const float*         c0      = (const float*)d_in[4];
  const float*         emb     = (const float*)d_in[5];
  const float*         W_ih    = (const float*)d_in[6];
  const float*         W_hh    = (const float*)d_in[7];
  const float*         W_in    = (const float*)d_in[8];
  const float*         W_out   = (const float*)d_in[9];
  const float*         W_lm    = (const float*)d_in[10];
  const float*         b_lm    = (const float*)d_in[11];
  float* out = (float*)d_out;

  // Workspace (~68 MB)
  char* w = (char*)d_ws;
  float* hseq  = (float*)w;  w += (size_t)ROWS * HID * 4;          // all h_t, f32
  float* c     = (float*)w;  w += (size_t)BSZ * HID * 4;           // cell state
  float* Q     = (float*)w;  w += (size_t)ROWS * ENCD * 4;         // q, f32
  h16*   czh   = (h16*)w;    w += (size_t)ROWS * (ENCD + HID) * 2; // [ctx|h], f16
  h16*   outs  = (h16*)w;    w += (size_t)ROWS * HID * 2;          // outputs, f16
  h16*   Wg16  = (h16*)w;    w += (size_t)2048 * 1024 * 2;         // [W_ih|W_hh] f16
  h16*   Win16 = (h16*)w;    w += (size_t)ENCD * HID * 2;
  h16*   Wout16= (h16*)w;    w += (size_t)HID * 1024 * 2;
  h16*   Wlm16 = (h16*)w;                                          // 32000x512 f16

  // Weight conversions (off the critical path, overlappable in HW)
  k_cvt_gates<<<2048, 256, 0, stream>>>(W_ih, W_hh, Wg16);
  k_cvt<<<512,  256, 0, stream>>>(W_in,  Win16,  ENCD * HID);
  k_cvt<<<1024, 256, 0, stream>>>(W_out, Wout16, HID * 1024);
  k_cvt<<<4096, 256, 0, stream>>>(W_lm,  Wlm16,  VOC * HID);

  hipMemcpyAsync(c, c0, (size_t)BSZ * HID * 4, hipMemcpyDeviceToDevice, stream);

  // Sequential recurrence: only the LSTM cell is on the critical path.
  const float* hprev = h0;
  for (int t = 0; t < NSTEP; ++t) {
    float* hcur = hseq + (size_t)t * BSZ * HID;
    k_gates<<<HID / 64, 256, 0, stream>>>(targets + t * BSZ, emb, Wg16,
                                          hprev, hcur, c);
    hprev = hcur;
  }

  // Batched: Q = H @ W_in^T   (4032 x 512 x 512)
  k_gemm<512, false, 0, 128><<<dim3(ENCD / 128, NSTEP), 256, 0, stream>>>(
      hseq, Win16, nullptr, Q, ENCD);

  // Batched attention -> [ctx | h] f16
  k_attn<<<dim3(BSZ, NSTEP), 128, 0, stream>>>(Q, enc, mask, hseq, czh);

  // Batched: out = tanh([ctx|h] @ W_out^T)  (4032 x 512 x 1024) -> f16
  k_gemm<1024, true, 1, 128><<<dim3(HID / 128, NSTEP), 256, 0, stream>>>(
      czh, Wout16, nullptr, outs, HID);

  // LM head: logits = outs @ W_lm^T + b  (4032 x 32000 x 512)
  k_gemm<512, true, 2, 256><<<dim3(VOC / 256, NSTEP), 256, 0, stream>>>(
      outs, Wlm16, b_lm, out, VOC);
}
